// MultiHeadAttention_18202071401083
// MI455X (gfx1250) — matrix operand order repaired and checked by reading, not compiled, not run on hardware
//
#include <hip/hip_runtime.h>
#include <hip/hip_bf16.h>

typedef __bf16 bf16_t;
typedef __attribute__((ext_vector_type(16))) __bf16 v16bf;
typedef __attribute__((ext_vector_type(8)))  __bf16 v8bf;
typedef __attribute__((ext_vector_type(8)))  float   v8f;
typedef __attribute__((ext_vector_type(4)))  float   v4f;

#define BATCH 2
#define NQTOK 2048
#define NKTOK 2048
#define DMODEL 1024
#define NHEAD 16
#define DHEAD 64

// ---------------------------------------------------------------------------
// WMMA helpers (CDNA5 gfx1250: 16x16x32 bf16 -> f32 accumulate)
// ---------------------------------------------------------------------------
__device__ __forceinline__ v8f wmma_bf16(v16bf a, v16bf b, v8f c) {
  return __builtin_amdgcn_wmma_f32_16x16x32_bf16(
      /*neg_a=*/false, a, /*neg_b=*/false, b,
      /*c_mod=*/(short)0, c, /*reuse_a=*/false, /*reuse_b=*/false);
}

// A-fragment (16x32, MxK): lane l holds row m=l&15; with hi=l>>4 the 16 bf16
// elements are K = hi*8 + [0..7] and K = 16 + hi*8 + [0..7]  (two b128 reads).
__device__ __forceinline__ v16bf load_a_frag(const bf16_t* base, int lda, int lane) {
  const int m = lane & 15, hi = lane >> 4;
  const bf16_t* p = base + m * lda + hi * 8;
  v8bf lo = *(const v8bf*)p;
  v8bf hb = *(const v8bf*)(p + 16);
  v16bf a;
#pragma unroll
  for (int i = 0; i < 8; ++i) { a[i] = lo[i]; a[i + 8] = hb[i]; }
  return a;
}

// B-fragment (32x16, KxN): lane l holds column n=l&15; with hi=l>>4 the 16 bf16
// elements are K = hi*8 + [0..7] and K = 16 + hi*8 + [0..7].
__device__ __forceinline__ v16bf load_b_frag(const bf16_t* base, int ldb, int lane) {
  const int n = lane & 15, hi = lane >> 4;
  v16bf b;
#pragma unroll
  for (int i = 0; i < 8; ++i) { b[i] = base[(hi * 8 + i) * ldb + n]; b[i + 8] = base[(16 + hi * 8 + i) * ldb + n]; }
  return b;
}

// ---------------------------------------------------------------------------
// Generic GEMM:  Y = (A[M,K] @ W[N,K]^T + bias) * out_scale
// K-step 64, software-pipelined global->reg->LDS staging.
// OUT_MODE 0: bf16 -> out[((b*16+h)*2048+q)*64 + d]        (Q / V layout)
// OUT_MODE 1: bf16 -> out[((b*16+h)*64+d)*2048 + q]        (K transposed)
// OUT_MODE 2: f32  -> out[r*N + c]                          (final output)
// ---------------------------------------------------------------------------
template <int OUT_MODE, bool A_BF16>
__global__ __launch_bounds__(256)
void gemm_kernel(const void* __restrict__ Ap, const float* __restrict__ W,
                 const float* __restrict__ bias, void* __restrict__ outp,
                 int M, int N, int K, float out_scale) {
  __shared__ __align__(32) bf16_t lds_a[128 * 72];   // [m][k], LDA=72 (K-step 64)
  __shared__ __align__(32) bf16_t lds_b[64 * 144];   // [k][n], LDB=144

  const int tid  = threadIdx.x;
  const int lane = tid & 31;
  const int wave = tid >> 5;
  const int wm   = wave & 3;        // 4 waves along M  (32 rows each)
  const int wn   = wave >> 2;       // 2 waves along N  (64 cols each)
  const int bm   = blockIdx.y * 128;
  const int bn   = blockIdx.x * 128;

  // staging geometry: each thread owns 32 contiguous K-elements of one row
  const int arow = tid >> 1, acol = (tid & 1) * 32;   // A tile 128x64
  const int brow = tid >> 1, bcol = (tid & 1) * 32;   // W rows (n) x 64 k

  v8f acc[2][4];
#pragma unroll
  for (int i = 0; i < 2; ++i)
#pragma unroll
    for (int j = 0; j < 4; ++j) acc[i][j] = (v8f)0.0f;

  v8bf areg[4], breg[4];

  // ---- prefetch helper ----
#define PREFETCH_TILES(k0_)                                                    \
  do {                                                                         \
    if (A_BF16) {                                                              \
      const bf16_t* g = (const bf16_t*)Ap + (size_t)(bm + arow) * K + (k0_) + acol; \
      _Pragma("unroll") for (int i = 0; i < 4; ++i)                            \
        areg[i] = *(const v8bf*)(g + 8 * i);                                   \
    } else {                                                                   \
      const float* g = (const float*)Ap + (size_t)(bm + arow) * K + (k0_) + acol; \
      v4f f[8];                                                                \
      _Pragma("unroll") for (int i = 0; i < 8; ++i) f[i] = *(const v4f*)(g + 4 * i); \
      _Pragma("unroll") for (int i = 0; i < 4; ++i)                            \
        _Pragma("unroll") for (int j = 0; j < 4; ++j) {                        \
          areg[i][j]     = (bf16_t)f[2 * i][j];                                \
          areg[i][j + 4] = (bf16_t)f[2 * i + 1][j];                            \
        }                                                                      \
    }                                                                          \
    {                                                                          \
      const float* g = W + (size_t)(bn + brow) * K + (k0_) + bcol;             \
      v4f f[8];                                                                \
      _Pragma("unroll") for (int i = 0; i < 8; ++i) f[i] = *(const v4f*)(g + 4 * i); \
      _Pragma("unroll") for (int i = 0; i < 4; ++i)                            \
        _Pragma("unroll") for (int j = 0; j < 4; ++j) {                        \
          breg[i][j]     = (bf16_t)f[2 * i][j];                                \
          breg[i][j + 4] = (bf16_t)f[2 * i + 1][j];                            \
        }                                                                      \
    }                                                                          \
  } while (0)

  PREFETCH_TILES(0);

  for (int k0 = 0; k0 < K; k0 += 64) {
    __syncthreads();
    // ---- commit staged registers to LDS ----
#pragma unroll
    for (int i = 0; i < 4; ++i)
      *(v8bf*)&lds_a[arow * 72 + acol + 8 * i] = areg[i];
#pragma unroll
    for (int i = 0; i < 4; ++i)
#pragma unroll
      for (int j = 0; j < 8; ++j)
        lds_b[(bcol + 8 * i + j) * 144 + brow] = breg[i][j];
    __syncthreads();

    if (k0 + 64 < K) PREFETCH_TILES(k0 + 64);   // overlap with WMMAs below

    // ---- 16 WMMAs per stage; all fragments hoisted ahead of the chain ----
#pragma unroll
    for (int ks = 0; ks < 64; ks += 32) {
      v16bf afr[2], bfr[4];
#pragma unroll
      for (int i = 0; i < 2; ++i)
        afr[i] = load_a_frag(&lds_a[(wm * 32 + i * 16) * 72 + ks], 72, lane);
#pragma unroll
      for (int j = 0; j < 4; ++j)
        bfr[j] = load_b_frag(&lds_b[ks * 144 + wn * 64 + j * 16], 144, lane);
#pragma unroll
      for (int i = 0; i < 2; ++i)
#pragma unroll
        for (int j = 0; j < 4; ++j)
          acc[i][j] = wmma_bf16(afr[i], bfr[j], acc[i][j]);
    }
  }
#undef PREFETCH_TILES

  // ---- epilogue: C layout (lane&15 = N col, VGPR v -> M = v + (lane>>4)*8) ----
  const int nl = lane & 15, hi = lane >> 4;
#pragma unroll
  for (int i = 0; i < 2; ++i) {
#pragma unroll
    for (int j = 0; j < 4; ++j) {
      const int c  = bn + wn * 64 + j * 16 + nl;
      const float bv = bias[c];
#pragma unroll
      for (int v = 0; v < 8; ++v) {
        const int r = bm + wm * 32 + i * 16 + v + hi * 8;
        const float val = (acc[i][j][v] + bv) * out_scale;
        if (OUT_MODE == 2) {
          ((float*)outp)[(size_t)r * N + c] = val;
        } else {
          const int b = r >> 11, q = r & 2047;
          const int h = c >> 6,  d = c & 63;
          if (OUT_MODE == 0)
            ((bf16_t*)outp)[(((size_t)(b * NHEAD + h) * NKTOK) + q) * DHEAD + d] = (bf16_t)val;
          else
            ((bf16_t*)outp)[(((size_t)(b * NHEAD + h) * DHEAD) + d) * NKTOK + q] = (bf16_t)val;
        }
      }
    }
  }
}

// ---------------------------------------------------------------------------
// Flash attention: one block = (b,h) x 128 query rows; 8 waves x 16 rows.
// Qb [bh][2048][64] (pre-scaled by 1/8), Kt [bh][64][2048], Vb [bh][2048][64].
// Ob out: bf16 [b][q][h*64+d]  (row-major [4096][1024] for the final GEMM).
// ---------------------------------------------------------------------------
__global__ __launch_bounds__(256)
void attn_kernel(const bf16_t* __restrict__ Qb, const bf16_t* __restrict__ Kt,
                 const bf16_t* __restrict__ Vb, const int* __restrict__ mask,
                 const float* __restrict__ bias, bf16_t* __restrict__ Ob) {
  __shared__ __align__(32) bf16_t lds_q[128 * 72];      // [q][d],   LDQ=72
  __shared__ __align__(32) bf16_t ktile[64 * 72];       // [d][key], LDK=72
  __shared__ __align__(32) bf16_t vtile[64 * 72];       // [key][d], LDV=72
  __shared__ __align__(32) bf16_t ptile[8 * 16 * 72];   // per-wave [m][key]
  __shared__ float maskadd[64];

  const int tid  = threadIdx.x;
  const int lane = tid & 31;
  const int wave = tid >> 5;
  const int bh   = blockIdx.y;                 // 0..31
  const int b    = bh >> 4, h = bh & 15;
  const int qbas = blockIdx.x * 128;
  const int q0   = qbas + wave * 16;
  const int nl   = lane & 15, hi = lane >> 4;

  // per-thread staging geometry for the 64x64 K/V tiles (16 elements each)
  const int krow = tid >> 2, kc0 = (tid & 3) * 16;

  // stage this block's 128 Q rows
  {
    const int row = tid >> 1, c0 = (tid & 1) * 32;
    const bf16_t* g = Qb + ((size_t)bh * NKTOK + qbas + row) * DHEAD + c0;
#pragma unroll
    for (int j = 0; j < 32; j += 8)
      *(v8bf*)&lds_q[row * 72 + c0 + j] = *(const v8bf*)(g + j);
  }

  v8f o[4];
#pragma unroll
  for (int t = 0; t < 4; ++t) o[t] = (v8f)0.0f;
  float mrow[8], lrow[8];
#pragma unroll
  for (int v = 0; v < 8; ++v) { mrow[v] = -1e30f; lrow[v] = 0.0f; }

  v8bf kreg[2], vreg[2];
  int mreg = 0;

#define PREFETCH_KV(kb_)                                                       \
  do {                                                                         \
    const bf16_t* gk = Kt + ((size_t)bh * DHEAD + krow) * NKTOK + (kb_) + kc0; \
    kreg[0] = *(const v8bf*)gk;                                                \
    kreg[1] = *(const v8bf*)(gk + 8);                                          \
    const bf16_t* gv = Vb + ((size_t)bh * NKTOK + (kb_) + krow) * DHEAD + kc0; \
    vreg[0] = *(const v8bf*)gv;                                                \
    vreg[1] = *(const v8bf*)(gv + 8);                                          \
    if (tid < 64) mreg = mask[b * NKTOK + (kb_) + tid];                        \
  } while (0)

  PREFETCH_KV(0);

  for (int kb = 0; kb < NKTOK; kb += 64) {
    __syncthreads();
    *(v8bf*)&ktile[krow * 72 + kc0]     = kreg[0];
    *(v8bf*)&ktile[krow * 72 + kc0 + 8] = kreg[1];
    *(v8bf*)&vtile[krow * 72 + kc0]     = vreg[0];
    *(v8bf*)&vtile[krow * 72 + kc0 + 8] = vreg[1];
    if (tid < 64) maskadd[tid] = mreg ? 0.0f : -1e9f;
    __syncthreads();

    if (kb + 64 < NKTOK) PREFETCH_KV(kb + 64);   // overlap with compute

    // ---- S accumulator initialized with bias + mask (C-operand of WMMA) ----
    v8f s[4];
#pragma unroll
    for (int v = 0; v < 8; ++v) {
      const int q = q0 + v + hi * 8;
      const float* brow = bias + ((size_t)b * NQTOK + q) * NKTOK + kb;
#pragma unroll
      for (int t = 0; t < 4; ++t)
        s[t][v] = brow[t * 16 + nl] + maskadd[t * 16 + nl];
    }

    // ---- S += Q . K^T  (Q pre-scaled by 1/8; dk=64 -> two 32-steps) ----
#pragma unroll
    for (int ks = 0; ks < 64; ks += 32) {
      v16bf aq = load_a_frag(&lds_q[(wave * 16) * 72 + ks], 72, lane);
      v16bf bk[4];
#pragma unroll
      for (int t = 0; t < 4; ++t)
        bk[t] = load_b_frag(&ktile[ks * 72 + t * 16], 72, lane);
#pragma unroll
      for (int t = 0; t < 4; ++t)
        s[t] = wmma_bf16(aq, bk[t], s[t]);
    }

    // ---- online softmax ----
#pragma unroll
    for (int v = 0; v < 8; ++v) {
      float vmax = -1e30f;
#pragma unroll
      for (int t = 0; t < 4; ++t) vmax = fmaxf(vmax, s[t][v]);
#pragma unroll
      for (int off = 1; off < 16; off <<= 1)
        vmax = fmaxf(vmax, __shfl_xor(vmax, off, 32));
      const float mnew  = fmaxf(mrow[v], vmax);
      const float alpha = __expf(mrow[v] - mnew);
      mrow[v] = mnew;
      float rsum = 0.0f;
#pragma unroll
      for (int t = 0; t < 4; ++t) {
        const float p = __expf(s[t][v] - mnew);
        s[t][v] = p;
        rsum += p;
      }
#pragma unroll
      for (int off = 1; off < 16; off <<= 1)
        rsum += __shfl_xor(rsum, off, 32);
      lrow[v] = lrow[v] * alpha + rsum;
#pragma unroll
      for (int t = 0; t < 4; ++t) o[t][v] *= alpha;
      // spill P row to this wave's private LDS tile (A-fragment source)
      const int m = v + hi * 8;
#pragma unroll
      for (int t = 0; t < 4; ++t)
        ptile[wave * 16 * 72 + m * 72 + t * 16 + nl] = (bf16_t)s[t][v];
    }

    // ---- O += P . V  (K-dim = 64 keys -> two 32-steps) ----
#pragma unroll
    for (int ks = 0; ks < 64; ks += 32) {
      v16bf ap = load_a_frag(&ptile[wave * 16 * 72 + ks], 72, lane);
      v16bf bv[4];
#pragma unroll
      for (int t = 0; t < 4; ++t)
        bv[t] = load_b_frag(&vtile[ks * 72 + t * 16], 72, lane);
#pragma unroll
      for (int t = 0; t < 4; ++t)
        o[t] = wmma_bf16(ap, bv[t], o[t]);
    }
  }
#undef PREFETCH_KV

  // ---- normalize and emit [b][q][h*64 + d] bf16 ----
#pragma unroll
  for (int v = 0; v < 8; ++v) {
    const float inv = 1.0f / lrow[v];
    const int q = q0 + v + hi * 8;
#pragma unroll
    for (int t = 0; t < 4; ++t)
      Ob[((size_t)b * NQTOK + q) * DMODEL + h * DHEAD + t * 16 + nl] =
          (bf16_t)(o[t][v] * inv);
  }
}

// ---------------------------------------------------------------------------
extern "C" void kernel_launch(void* const* d_in, const int* in_sizes, int n_in,
                              void* d_out, int out_size, void* d_ws, size_t ws_size,
                              hipStream_t stream) {
  (void)in_sizes; (void)n_in; (void)out_size; (void)ws_size;
  const float* query = (const float*)d_in[0];
  const float* key   = (const float*)d_in[1];
  const float* value = (const float*)d_in[2];
  const int*   mask  = (const int*)  d_in[3];
  const float* abias = (const float*)d_in[4];
  const float* w_q   = (const float*)d_in[5];
  const float* b_q   = (const float*)d_in[6];
  const float* w_k   = (const float*)d_in[7];
  const float* b_k   = (const float*)d_in[8];
  const float* w_v   = (const float*)d_in[9];
  const float* b_v   = (const float*)d_in[10];
  const float* w_o   = (const float*)d_in[11];
  const float* b_o   = (const float*)d_in[12];

  const size_t NE = (size_t)BATCH * NHEAD * NKTOK * DHEAD;  // 4M elements
  bf16_t* Qb = (bf16_t*)d_ws;
  bf16_t* Kt = Qb + NE;
  bf16_t* Vb = Kt + NE;
  bf16_t* Ob = Vb + NE;

  const dim3 blk(256);
  const dim3 gproj(DMODEL / 128, (BATCH * NQTOK) / 128);  // (8, 32)
  // Q projection carries the 1/sqrt(dk) scale: (x@Wq^T + bq) * 0.125
  gemm_kernel<0, false><<<gproj, blk, 0, stream>>>(query, w_q, b_q, Qb,
                                                   BATCH * NQTOK, DMODEL, DMODEL, 0.125f);
  gemm_kernel<1, false><<<gproj, blk, 0, stream>>>(key, w_k, b_k, Kt,
                                                   BATCH * NKTOK, DMODEL, DMODEL, 1.0f);
  gemm_kernel<0, false><<<gproj, blk, 0, stream>>>(value, w_v, b_v, Vb,
                                                   BATCH * NKTOK, DMODEL, DMODEL, 1.0f);

  const dim3 gattn(NQTOK / 128, BATCH * NHEAD);           // (16, 32)
  attn_kernel<<<gattn, blk, 0, stream>>>(Qb, Kt, Vb, mask, abias, Ob);

  gemm_kernel<2, true><<<gproj, blk, 0, stream>>>(Ob, w_o, b_o, (float*)d_out,
                                                  BATCH * NQTOK, DMODEL, DMODEL, 1.0f);
}
